// DGCNNDecoder_3126736192094
// MI455X (gfx1250) — matrix operand order repaired and checked by reading, not compiled, not run on hardware
//
#include <hip/hip_runtime.h>

typedef __attribute__((ext_vector_type(16))) __bf16 v16bf;
typedef __attribute__((ext_vector_type(8)))  __bf16 v8bf;
typedef __attribute__((ext_vector_type(8)))  float  v8f;

constexpr int Bc    = 2;
constexpr int Nc    = 2048;
constexpr int NYc   = 4096;
constexpr int NOBJc = 4;
constexpr int CDIMc = 32;
constexpr int Hc    = 256;
constexpr int Kc    = 20;
constexpr int NBLK  = 5;
constexpr int Mc    = Bc * NYc;          // 8192 pc points
constexpr int NPOS1 = Mc * Kc;           // 163840 edge positions
constexpr int NPOS2 = Bc * NOBJc * Nc;   // 16384 decoder positions
constexpr float EPS = 1e-5f;

__device__ __forceinline__ __bf16 to_bf16(float x) { return (__bf16)x; }

// ---------------------------------------------------------------------------
// Weight conversion kernels (fp32 -> bf16, with zero-pad or transpose)
// ---------------------------------------------------------------------------
__global__ void cvt_pad_kernel(const float* __restrict__ in, __bf16* __restrict__ out,
                               int Cin, int CinP, int total) {
    int idx = blockIdx.x * 256 + threadIdx.x;
    if (idx >= total) return;
    int o = idx / CinP, c = idx % CinP;
    out[idx] = (c < Cin) ? to_bf16(in[o * Cin + c]) : to_bf16(0.0f);
}

__global__ void cvt_T_kernel(const float* __restrict__ in, __bf16* __restrict__ out,
                             int Cout, int Cin, int total) {
    int idx = blockIdx.x * 256 + threadIdx.x;
    if (idx >= total) return;
    int o = idx / Cin, c = idx % Cin;
    out[idx] = to_bf16(in[c * Cout + o]);   // in is (Cin, Cout) -> out (Cout, Cin)
}

__global__ void bnfold_kernel(const float* __restrict__ gamma, const float* __restrict__ beta,
                              const float* __restrict__ mean,  const float* __restrict__ var,
                              float* __restrict__ scale, float* __restrict__ shift, int C) {
    int i = blockIdx.x * 256 + threadIdx.x;
    if (i >= C) return;
    float s = gamma[i] * rsqrtf(var[i] + EPS);
    scale[i] = s;
    shift[i] = beta[i] - mean[i] * s;
}

// ---------------------------------------------------------------------------
// KNN + edge-feature build: for each pc point, top-K nearest candidates of
// p[node_tag % B]; emit 64-channel (38 used) bf16 rows, position-major.
// ---------------------------------------------------------------------------
__global__ __launch_bounds__(128)
void knn_build_kernel(const float* __restrict__ p, const float* __restrict__ pc,
                      const float* __restrict__ feat, const int* __restrict__ node_tag,
                      __bf16* __restrict__ X0) {
    __shared__ float sp[Bc * Nc * 3];   // 48 KB: all candidate points
    for (int i = threadIdx.x; i < Bc * Nc * 3; i += 128) sp[i] = p[i];
    __syncthreads();

    const int m = blockIdx.x * 128 + threadIdx.x;   // pc point index, M = 8192
    const int cb = node_tag[m] % Bc;
    const float* cp = &sp[cb * Nc * 3];

    const float yx = pc[m * 3 + 0], yy = pc[m * 3 + 1], yz = pc[m * 3 + 2];

    float bd[Kc]; int bi[Kc];
#pragma unroll
    for (int k = 0; k < Kc; k++) { bd[k] = 3.0e38f; bi[k] = 0; }

    for (int j = 0; j < Nc; j++) {
        float dx = yx - cp[j * 3 + 0];
        float dy = yy - cp[j * 3 + 1];
        float dz = yz - cp[j * 3 + 2];
        float d = dx * dx + dy * dy + dz * dz;
        if (d < bd[Kc - 1]) {
            int t = Kc - 1;
            while (t > 0 && bd[t - 1] > d) { bd[t] = bd[t - 1]; bi[t] = bi[t - 1]; t--; }
            bd[t] = d; bi[t] = j;
        }
    }

    __bf16 fb[CDIMc];
    const float* fr = feat + (size_t)m * CDIMc;
#pragma unroll
    for (int c = 0; c < CDIMc; c++) fb[c] = to_bf16(fr[c]);

    for (int k = 0; k < Kc; k++) {
        int j = bi[k];
        float x0 = cp[j * 3 + 0], x1 = cp[j * 3 + 1], x2 = cp[j * 3 + 2];
        __bf16* o = X0 + (size_t)(m * Kc + k) * 64;
        o[0] = to_bf16(yx - x0); o[1] = to_bf16(yy - x1); o[2] = to_bf16(yz - x2);
        o[3] = to_bf16(x0);      o[4] = to_bf16(x1);      o[5] = to_bf16(x2);
#pragma unroll
        for (int c = 0; c < CDIMc; c++) o[6 + c] = fb[c];
#pragma unroll
        for (int c = 38; c < 64; c++) o[c] = to_bf16(0.0f);
    }
}

// ---------------------------------------------------------------------------
// WMMA fragment builders (layouts per CDNA5 ISA 7.12.2, wave32)
// A (16x32 bf16): elems 0..7 -> K = kk+8g..,  elems 8..15 -> K = kk+16+8g..
// B (32x16 bf16): same as A: elems 0..7 -> K = kk+8g..,  elems 8..15 -> K = kk+16+8g..
// ---------------------------------------------------------------------------
__device__ __forceinline__ v16bf load_a_frag(const __bf16* __restrict__ row, int kk, int g) {
    v8bf a0 = *(const v8bf*)(row + kk + 8 * g);
    v8bf a1 = *(const v8bf*)(row + kk + 16 + 8 * g);
    v16bf A;
#pragma unroll
    for (int i = 0; i < 8; i++) { A[i] = a0[i]; A[i + 8] = a1[i]; }
    return A;
}

__device__ __forceinline__ v16bf load_b_frag(const __bf16* __restrict__ row, int kk, int g) {
    v8bf b0 = *(const v8bf*)(row + kk + 8 * g);
    v8bf b1 = *(const v8bf*)(row + kk + 16 + 8 * g);
    v16bf B;
#pragma unroll
    for (int i = 0; i < 8; i++) { B[i] = b0[i]; B[i + 8] = b1[i]; }
    return B;
}

// ---------------------------------------------------------------------------
// Register-blocked bf16 WMMA GEMM: Out[pos][cout] = sum_c W[cout][c]*X[pos][c]
// Wave computes a 32(cout) x 64(pos) block = 2x4 tiles of 16x16:
//   per K-step: 2 A-frags + 4 B-frags (12 x b128) feed 8 WMMAs.
// Block = 4 waves -> 32 couts x 256 positions.
//   POST 0: BN scale/shift + LeakyReLU(0.2) -> bf16
//   POST 1: BN scale/shift + LeakyReLU(0.2) -> f32
//   POST 2: +bias, ReLU -> bf16
//   POST 3: +bias, residual add into f32 buffer
// ---------------------------------------------------------------------------
template<int POST>
__global__ __launch_bounds__(128)
void gemm32_kernel(const __bf16* __restrict__ W, const __bf16* __restrict__ X,
                   int Cin, int Cout,
                   const float* __restrict__ s0, const float* __restrict__ s1,
                   float* __restrict__ outF, __bf16* __restrict__ outB) {
    const int wave = threadIdx.x >> 5;
    const int lane = threadIdx.x & 31;
    const int r = lane & 15;           // A row / B column within tile
    const int g = lane >> 4;           // lane-group select
    const size_t p0 = (size_t)blockIdx.x * 256 + wave * 64;   // 4 pos tiles
    const int m0 = blockIdx.y * 32;                           // 2 cout tiles

    const __bf16* wrow[2];
    const __bf16* xrow[4];
#pragma unroll
    for (int i = 0; i < 2; i++) wrow[i] = W + (size_t)(m0 + 16 * i + r) * Cin;
#pragma unroll
    for (int j = 0; j < 4; j++) xrow[j] = X + (p0 + 16 * j + r) * (size_t)Cin;

    v8f acc[2][4];
#pragma unroll
    for (int i = 0; i < 2; i++)
#pragma unroll
        for (int j = 0; j < 4; j++) acc[i][j] = (v8f){};

    for (int kk = 0; kk < Cin; kk += 32) {
        v16bf A[2], Bm[4];
#pragma unroll
        for (int i = 0; i < 2; i++) A[i] = load_a_frag(wrow[i], kk, g);
#pragma unroll
        for (int j = 0; j < 4; j++) Bm[j] = load_b_frag(xrow[j], kk, g);
#pragma unroll
        for (int i = 0; i < 2; i++)
#pragma unroll
            for (int j = 0; j < 4; j++)
                acc[i][j] = __builtin_amdgcn_wmma_f32_16x16x32_bf16(
                    false, A[i], false, Bm[j], (short)0, acc[i][j], false, false);
    }

#pragma unroll
    for (int i = 0; i < 2; i++) {
#pragma unroll
        for (int j = 0; j < 4; j++) {
#pragma unroll
            for (int v = 0; v < 8; v++) {
                const int cout = m0 + 16 * i + v + 8 * g;   // D: VGPR v -> M=v (+8 hi lanes)
                const size_t pos = p0 + 16 * j + r;         // D: N = lane & 15
                const size_t oi = pos * (size_t)Cout + cout;
                float d = acc[i][j][v];
                if constexpr (POST == 0 || POST == 1) {
                    d = d * s0[cout] + s1[cout];
                    d = d > 0.0f ? d : 0.2f * d;
                    if constexpr (POST == 0) outB[oi] = to_bf16(d);
                    else                     outF[oi] = d;
                } else if constexpr (POST == 2) {
                    d += s0[cout];
                    d = d > 0.0f ? d : 0.0f;
                    outB[oi] = to_bf16(d);
                } else {
                    d += s0[cout];
                    outF[oi] = outF[oi] + d;
                }
            }
        }
    }
}

// ---------------------------------------------------------------------------
// Reductions and small dense ops
// ---------------------------------------------------------------------------
__global__ void maxk_kernel(const float* __restrict__ Y3, float* __restrict__ c) {
    int idx = blockIdx.x * 256 + threadIdx.x;      // M*CDIM = 262144
    if (idx >= Mc * CDIMc) return;
    int m = idx >> 5, ch = idx & 31;
    float v = -3.0e38f;
#pragma unroll
    for (int k = 0; k < Kc; k++)
        v = fmaxf(v, Y3[(size_t)(m * Kc + k) * CDIMc + ch]);
    c[idx] = v;
}

__global__ void segmax_kernel(const float* __restrict__ c, const int* __restrict__ node_tag,
                              float* __restrict__ obj) {
    int t = threadIdx.x;                           // 256 = 8 segs * 32 ch
    int tag = t >> 5, ch = t & 31;
    float v = -3.0e38f;
    for (int m = 0; m < Mc; m++)
        if (node_tag[m] == tag) v = fmaxf(v, c[m * CDIMc + ch]);
    obj[t] = v;
}

__global__ void net0_kernel(const float* __restrict__ p, const float* __restrict__ w,
                            const float* __restrict__ b, float* __restrict__ net) {
    int idx = blockIdx.x * 256 + threadIdx.x;      // NPOS2 * H
    if (idx >= NPOS2 * Hc) return;
    int q = idx >> 8, h = idx & 255;
    int row = q >> 11;                             // q / N -> b*NOBJ+o
    int bb = row >> 2;                             // / NOBJ
    int n = q & (Nc - 1);
    const float* pp = p + (size_t)(bb * Nc + n) * 3;
    net[idx] = pp[0] * w[h] + pp[1] * w[Hc + h] + pp[2] * w[2 * Hc + h] + b[h];
}

__global__ void cc_kernel(const float* __restrict__ obj, const float* __restrict__ w,
                          const float* __restrict__ b, float* __restrict__ cc) {
    int h = threadIdx.x;                           // 256 threads, 1 block
    for (int o = 0; o < Bc * NOBJc; o++) {
        float s = b[h];
#pragma unroll
        for (int c = 0; c < CDIMc; c++) s += obj[o * CDIMc + c] * w[c * Hc + h];
        cc[o * Hc + h] = s;
    }
}

__global__ void addcc_kernel(float* __restrict__ net, const float* __restrict__ cc,
                             __bf16* __restrict__ Xr) {
    int idx = blockIdx.x * 256 + threadIdx.x;
    if (idx >= NPOS2 * Hc) return;
    int q = idx >> 8, h = idx & 255;
    int row = q >> 11;
    float v = net[idx] + cc[row * Hc + h];
    net[idx] = v;
    Xr[idx] = to_bf16(v > 0.0f ? v : 0.0f);
}

__global__ void out_kernel(const float* __restrict__ net, const float* __restrict__ w,
                           const float* __restrict__ b, float* __restrict__ out) {
    int q = blockIdx.x * 256 + threadIdx.x;
    if (q >= NPOS2) return;
    float s = b[0];
    const float* nr = net + (size_t)q * Hc;
    for (int h = 0; h < Hc; h++) {
        float v = nr[h];
        s += (v > 0.0f ? v : 0.0f) * w[h];
    }
    out[q] = s;
}

// ---------------------------------------------------------------------------
// Host orchestration
// ---------------------------------------------------------------------------
extern "C" void kernel_launch(void* const* d_in, const int* in_sizes, int n_in,
                              void* d_out, int out_size, void* d_ws, size_t ws_size,
                              hipStream_t stream) {
    (void)in_sizes; (void)n_in; (void)out_size; (void)ws_size;

    // ---- inputs (jax pytree order: p, pc, feat, node_tag, then params leaves,
    //      dict keys sorted: blocks, bn1, bn2, bn3, conv1_w, conv2_w, conv3_w,
    //      fc_c, fc_out, fc_p; nested dicts also key-sorted: b before w, etc.)
    const float* p        = (const float*)d_in[0];
    const float* pc       = (const float*)d_in[1];
    const float* feat     = (const float*)d_in[2];
    const int*   node_tag = (const int*)d_in[3];

    const float *fc0_b[NBLK], *fc0_w[NBLK], *fc1_b[NBLK], *fc1_w[NBLK];
    for (int i = 0; i < NBLK; i++) {
        int base = 4 + 4 * i;
        fc0_b[i] = (const float*)d_in[base + 0];
        fc0_w[i] = (const float*)d_in[base + 1];
        fc1_b[i] = (const float*)d_in[base + 2];
        fc1_w[i] = (const float*)d_in[base + 3];
    }
    const float* bn1_beta = (const float*)d_in[24];
    const float* bn1_gam  = (const float*)d_in[25];
    const float* bn1_mean = (const float*)d_in[26];
    const float* bn1_var  = (const float*)d_in[27];
    const float* bn2_beta = (const float*)d_in[28];
    const float* bn2_gam  = (const float*)d_in[29];
    const float* bn2_mean = (const float*)d_in[30];
    const float* bn2_var  = (const float*)d_in[31];
    const float* bn3_beta = (const float*)d_in[32];
    const float* bn3_gam  = (const float*)d_in[33];
    const float* bn3_mean = (const float*)d_in[34];
    const float* bn3_var  = (const float*)d_in[35];
    const float* conv1_w  = (const float*)d_in[36];
    const float* conv2_w  = (const float*)d_in[37];
    const float* conv3_w  = (const float*)d_in[38];
    const float *fcc_b[NBLK], *fcc_w[NBLK];
    for (int i = 0; i < NBLK; i++) {
        fcc_b[i] = (const float*)d_in[39 + 2 * i];
        fcc_w[i] = (const float*)d_in[40 + 2 * i];
    }
    const float* fcout_b  = (const float*)d_in[49];
    const float* fcout_w  = (const float*)d_in[50];
    const float* fcp_b    = (const float*)d_in[51];
    const float* fcp_w    = (const float*)d_in[52];

    // ---- workspace carving
    char* ws = (char*)d_ws;
    size_t off = 0;
    auto carve = [&](size_t bytes) -> char* {
        char* r = ws + off;
        off = (off + bytes + 255) & ~(size_t)255;
        return r;
    };
    __bf16* Wc1  = (__bf16*)carve((size_t)Hc * 64 * 2);
    __bf16* Wc2  = (__bf16*)carve((size_t)Hc * Hc * 2);
    __bf16* Wc3  = (__bf16*)carve((size_t)CDIMc * Hc * 2);
    __bf16* Wfc  = (__bf16*)carve((size_t)2 * NBLK * Hc * Hc * 2);  // [2i]=fc0^T,[2i+1]=fc1^T
    float* bnS1 = (float*)carve(Hc * 4);    float* bnH1 = (float*)carve(Hc * 4);
    float* bnS2 = (float*)carve(Hc * 4);    float* bnH2 = (float*)carve(Hc * 4);
    float* bnS3 = (float*)carve(CDIMc * 4); float* bnH3 = (float*)carve(CDIMc * 4);
    float*  cbuf = (float*)carve((size_t)Mc * CDIMc * 4);
    float*  objb = (float*)carve((size_t)Bc * NOBJc * CDIMc * 4);
    float*  netb = (float*)carve((size_t)NPOS2 * Hc * 4);
    __bf16* Xr   = (__bf16*)carve((size_t)NPOS2 * Hc * 2);
    __bf16* Hr   = (__bf16*)carve((size_t)NPOS2 * Hc * 2);
    float*  ccb  = (float*)carve((size_t)Bc * NOBJc * Hc * 4);
    char* R1 = carve((size_t)NPOS1 * Hc * 2);   // 84 MB
    char* R2 = carve((size_t)NPOS1 * Hc * 2);   // 84 MB
    __bf16* X0 = (__bf16*)R2;                   // 163840 x 64 bf16 (first 21 MB of R2)
    __bf16* A1 = (__bf16*)R1;                   // conv1 out 163840 x 256 bf16
    __bf16* A2 = (__bf16*)R2;                   // conv2 out (X0 dead by then)
    float*  Y3 = (float*)R1;                    // conv3 out 163840 x 32 f32 (A1 dead)

    auto g1 = [](int total) { return dim3((total + 255) / 256); };

    // ---- weight prep
    cvt_pad_kernel<<<g1(Hc * 64), 256, 0, stream>>>(conv1_w, Wc1, 38, 64, Hc * 64);
    cvt_pad_kernel<<<g1(Hc * Hc), 256, 0, stream>>>(conv2_w, Wc2, Hc, Hc, Hc * Hc);
    cvt_pad_kernel<<<g1(CDIMc * Hc), 256, 0, stream>>>(conv3_w, Wc3, Hc, Hc, CDIMc * Hc);
    for (int i = 0; i < NBLK; i++) {
        cvt_T_kernel<<<g1(Hc * Hc), 256, 0, stream>>>(fc0_w[i], Wfc + (size_t)(2 * i) * Hc * Hc, Hc, Hc, Hc * Hc);
        cvt_T_kernel<<<g1(Hc * Hc), 256, 0, stream>>>(fc1_w[i], Wfc + (size_t)(2 * i + 1) * Hc * Hc, Hc, Hc, Hc * Hc);
    }
    bnfold_kernel<<<1, 256, 0, stream>>>(bn1_gam, bn1_beta, bn1_mean, bn1_var, bnS1, bnH1, Hc);
    bnfold_kernel<<<1, 256, 0, stream>>>(bn2_gam, bn2_beta, bn2_mean, bn2_var, bnS2, bnH2, Hc);
    bnfold_kernel<<<1, 256, 0, stream>>>(bn3_gam, bn3_beta, bn3_mean, bn3_var, bnS3, bnH3, CDIMc);

    // ---- KNN + edge features
    knn_build_kernel<<<Mc / 128, 128, 0, stream>>>(p, pc, feat, node_tag, X0);

    // ---- edge-conv GEMM chain (WMMA bf16, 32x256 block per workgroup)
    gemm32_kernel<0><<<dim3(NPOS1 / 256, Hc / 32), 128, 0, stream>>>(Wc1, X0, 64, Hc, bnS1, bnH1, nullptr, A1);
    gemm32_kernel<0><<<dim3(NPOS1 / 256, Hc / 32), 128, 0, stream>>>(Wc2, A1, Hc, Hc, bnS2, bnH2, nullptr, A2);
    gemm32_kernel<1><<<dim3(NPOS1 / 256, CDIMc / 32), 128, 0, stream>>>(Wc3, A2, Hc, CDIMc, bnS3, bnH3, Y3, nullptr);

    // ---- reductions
    maxk_kernel<<<g1(Mc * CDIMc), 256, 0, stream>>>(Y3, cbuf);
    segmax_kernel<<<1, 256, 0, stream>>>(cbuf, node_tag, objb);

    // ---- decoder
    net0_kernel<<<g1(NPOS2 * Hc), 256, 0, stream>>>(p, fcp_w, fcp_b, netb);
    for (int i = 0; i < NBLK; i++) {
        cc_kernel<<<1, 256, 0, stream>>>(objb, fcc_w[i], fcc_b[i], ccb);
        addcc_kernel<<<g1(NPOS2 * Hc), 256, 0, stream>>>(netb, ccb, Xr);
        gemm32_kernel<2><<<dim3(NPOS2 / 256, Hc / 32), 128, 0, stream>>>(
            Wfc + (size_t)(2 * i) * Hc * Hc, Xr, Hc, Hc, fc0_b[i], nullptr, nullptr, Hr);
        gemm32_kernel<3><<<dim3(NPOS2 / 256, Hc / 32), 128, 0, stream>>>(
            Wfc + (size_t)(2 * i + 1) * Hc * Hc, Hr, Hc, Hc, fc1_b[i], nullptr, netb, nullptr);
    }
    out_kernel<<<g1(NPOS2), 256, 0, stream>>>(netb, fcout_w, fcout_b, (float*)d_out);
}
